// Boundary_binaryLoss_22582938042733
// MI455X (gfx1250) — matrix operand order repaired and checked by reading, not compiled, not run on hardware
//
#include <hip/hip_runtime.h>

// Problem constants (from the reference)
#define Bv   32
#define Cv   2
#define Hv   480
#define Wv   864
#define Rv   7            // kernel radius (15x15)
#define IGN  255
#define TW   64           // tile width  (outputs per block, x)
#define TH   64           // tile height (outputs per block, y)
#define HWd  (TW + 2*Rv)  // 78: halo tile width
#define HHd  (TH + 2*Rv)  // 78: halo tile height (label tile rows)
#define SRH  80           // sRow rows (padded to 80 so B's K=30,31 reads are in-bounds)
#define NTHREADS 256

typedef __attribute__((ext_vector_type(16))) _Float16 v16h;
typedef __attribute__((ext_vector_type(8)))  float    v8f;

__global__ __launch_bounds__(NTHREADS)
void bboundary_init_kernel(float* gsum, int* gcnt) {
    if (threadIdx.x == 0) { gsum[0] = 0.0f; gcnt[0] = 0; }
}

__global__ __launch_bounds__(NTHREADS)
void bboundary_loss_kernel(const int* __restrict__ labels,
                           const float* __restrict__ logits,
                           float* __restrict__ gsum,
                           int* __restrict__ gcnt) {
    __shared__ int           sLbl[HHd * HWd];   // 78*78 int32 = 24336 B
    __shared__ unsigned char sRow[SRH * TW];    // 80*64 bytes =  5120 B
    __shared__ float         rs[NTHREADS];
    __shared__ int           rc[NTHREADS];

    const int tid = threadIdx.x;
    const int bx = blockIdx.x, by = blockIdx.y, b = blockIdx.z;
    const int gw0 = bx * TW - Rv;   // global col of LDS tile col 0
    const int gh0 = by * TH - Rv;   // global row of LDS tile row 0
    const long long lblBase = (long long)b * Hv * Wv;

    // ---- Stage 1: async-copy label halo tile global -> LDS (ASYNCcnt path) ----
    for (int i = tid; i < HHd * HWd; i += NTHREADS) {
        const int r = i / HWd, c = i - r * HWd;
        const int gh = gh0 + r, gw = gw0 + c;
        if (gh >= 0 && gh < Hv && gw >= 0 && gw < Wv) {
            const unsigned long long gaddr =
                (unsigned long long)(labels + lblBase + (long long)gh * Wv + gw);
            const unsigned ldsOff = (unsigned)(unsigned long long)(&sLbl[i]);
            asm volatile("global_load_async_to_lds_b32 %0, %1, off"
                         :: "v"(ldsOff), "v"(gaddr) : "memory");
        }
    }

    // Warm L2 with logits lines this block gathers later (both channels).
    {
        const int ty = tid >> 6, tx = tid & (TW - 1);
        const int h = by * TH + ty * (TH / 4);
        const int w = bx * TW + tx;
        if (h < Hv && w < Wv) {
            __builtin_prefetch(&logits[(((long long)b * Cv + 0) * Hv + h) * Wv + w], 0, 0);
            __builtin_prefetch(&logits[(((long long)b * Cv + 1) * Hv + h) * Wv + w], 0, 0);
        }
    }

    asm volatile("s_wait_asynccnt 0x0" ::: "memory");
    __syncthreads();

    // ---- Stage 2: horizontal clamped 15-count of (lbl==1) into sRow ----
    for (int i = tid; i < SRH * TW; i += NTHREADS) {
        const int r = i >> 6;            // sRow row (0..79)
        const int c = i & (TW - 1);      // output col within tile
        const int gh = gh0 + r;
        const int w  = bx * TW + c;
        int cnt = 0;
        if (r < HHd && gh >= 0 && gh < Hv && w < Wv) {
            const int lo = (w - Rv < 0)      ? -w           : -Rv;
            const int hi = (w + Rv > Wv - 1) ? (Wv - 1 - w) :  Rv;
            const int base = r * HWd + (c + Rv);
            for (int dw = lo; dw <= hi; ++dw)
                cnt += (sLbl[base + dw] == 1);
        }
        sRow[i] = (unsigned char)cnt;
    }
    __syncthreads();

    // ---- Stage 3: vertical clamped 15-sum as a banded WMMA matmul ----
    // Per 16x16 output subtile: S = A(16x32) x B(32x16), one v_wmma_f32_16x16x32_f16.
    // A[m][k] = 1 iff k in [m, m+14] and global input row (h0-7+k) is in-bounds.
    const int wave = tid >> 5;          // 8 waves
    const int L    = tid & 31;          // lane in wave (wave32)
    const bool lo16 = (L < 16);
    const int mn   = L & 15;            // A's M index == B/D's N index for this lane

    float lsum = 0.0f;
    int   lcnt = 0;

    for (int st = 2 * wave; st < 2 * wave + 2; ++st) {   // 16 subtiles / 8 waves
        const int sy = st >> 2, sx = st & 3;
        const int h0 = by * TH + sy * 16;
        const int w0 = bx * TW + sx * 16;

        // Build A fragment (band matrix + vertical border clamp), exact in f16.
        v16h a;
#pragma unroll
        for (int i = 0; i < 16; ++i) {
            const int k = i + (lo16 ? 0 : 16);  // K index of slot i
            const int j = h0 - Rv + k;                             // global input row
            const bool on = (k >= mn) && (k <= mn + 2 * Rv) && (j >= 0) && (j < Hv);
            a[i] = on ? (_Float16)1.0f : (_Float16)0.0f;
        }

        // Build B fragment: column (w0+n) of the 32-row sRow strip (A's k >= 30 columns are zero).
        v16h bmat;
        const int k0   = lo16 ? 0 : 16;
        const int colb = sx * 16 + mn;
#pragma unroll
        for (int i = 0; i < 16; ++i) {
            const int row = sy * 16 + k0 + i;                      // 0..79, in-bounds
            bmat[i] = (_Float16)(float)sRow[row * TW + colb];
        }

        v8f acc = {};
        acc = __builtin_amdgcn_wmma_f32_16x16x32_f16(
            /*neg_a=*/false, a, /*neg_b=*/false, bmat,
            /*c_mod=*/(short)0, acc, /*reuse_a=*/false, /*reuse_b=*/false);

        // Validity test + logit gather for this lane's 8 output pixels.
#pragma unroll
        for (int r = 0; r < 8; ++r) {
            const int m = r + (lo16 ? 0 : 8);
            const int h = h0 + m;
            const int w = w0 + mn;
            if (h < Hv && w < Wv) {
                const int S = (int)(acc[r] + 0.5f);   // exact small integer
                const int lbl = sLbl[(sy * 16 + m + Rv) * HWd + (sx * 16 + mn + Rv)];
                const int jlo = (h - Rv < 0)      ? 0      : h - Rv;
                const int jhi = (h + Rv > Hv - 1) ? Hv - 1 : h + Rv;
                const int wlo = (w - Rv < 0)      ? 0      : w - Rv;
                const int whi = (w + Rv > Wv - 1) ? Wv - 1 : w + Rv;
                const int N = (whi - wlo + 1) * (jhi - jlo + 1);
                // boundary <=> window holds both a 1 and a non-1 in-bounds pixel
                if ((lbl != IGN) && (S > 0) && (S < N)) {
                    lsum += logits[(((long long)b * Cv + lbl) * Hv + h) * Wv + w];
                    lcnt += 1;
                }
            }
        }
    }

    // Block tree reduction, then one atomic per block.
    rs[tid] = lsum;
    rc[tid] = lcnt;
    __syncthreads();
    for (int off = NTHREADS / 2; off > 0; off >>= 1) {
        if (tid < off) { rs[tid] += rs[tid + off]; rc[tid] += rc[tid + off]; }
        __syncthreads();
    }
    if (tid == 0) {
        atomicAdd(gsum, rs[0]);
        atomicAdd(gcnt, rc[0]);
    }
}

__global__ __launch_bounds__(NTHREADS)
void bboundary_finalize_kernel(const float* gsum, const int* gcnt, float* out) {
    if (threadIdx.x == 0) {
        const int n = (gcnt[0] < 1) ? 1 : gcnt[0];
        out[0] = -gsum[0] / (float)n;
    }
}

extern "C" void kernel_launch(void* const* d_in, const int* in_sizes, int n_in,
                              void* d_out, int out_size, void* d_ws, size_t ws_size,
                              hipStream_t stream) {
    (void)in_sizes; (void)n_in; (void)out_size; (void)ws_size;
    const float* logits = (const float*)d_in[0];   // [B, C, H, W] float32
    const int*   labels = (const int*)d_in[1];     // [B, H, W]    int32
    float* out  = (float*)d_out;                   // scalar float32
    float* gsum = (float*)d_ws;
    int*   gcnt = (int*)((char*)d_ws + sizeof(float));

    bboundary_init_kernel<<<1, NTHREADS, 0, stream>>>(gsum, gcnt);

    dim3 grid((Wv + TW - 1) / TW, (Hv + TH - 1) / TH, Bv);  // (14, 8, 32)
    bboundary_loss_kernel<<<grid, NTHREADS, 0, stream>>>(labels, logits, gsum, gcnt);

    bboundary_finalize_kernel<<<1, NTHREADS, 0, stream>>>(gsum, gcnt, out);
}
